// CrissCrossAttention_69063074120027
// MI455X (gfx1250) — matrix operand order repaired and checked by reading, not compiled, not run on hardware
//
#include <hip/hip_runtime.h>
#include <hip/hip_bf16.h>

typedef _Float16 f16;
typedef f16   v16h __attribute__((ext_vector_type(16)));
typedef float v8f  __attribute__((ext_vector_type(8)));

union Frag { v16h v; unsigned int u[8]; };

#define WMMA_F16(a, b, c) \
  __builtin_amdgcn_wmma_f32_16x16x32_f16(false, (a), false, (b), (short)0, (c), false, false)

// ---------------------------------------------------------------------------
// Async copy of a 64x64 f16 tile (rows contiguous in memory, row stride 4096
// elems) into LDS rows of `rowStrideLds` elems, via CDNA5 async-to-LDS DMA.
// Per-lane B32 granularity (2 f16). Tracked by ASYNCcnt.
// ---------------------------------------------------------------------------
__device__ __forceinline__ void async_copy_tile64(
    const f16* __restrict__ gsrc, f16* lds, int rowStrideLds, int tid) {
  for (int e = tid; e < 2048; e += 128) {          // 64 rows * 32 uints
    int r = e >> 5, c2 = (e & 31) << 1;
    unsigned      ldsOff = (unsigned)(unsigned long long)&lds[r * rowStrideLds + c2];
    unsigned long long ga = (unsigned long long)(gsrc + (size_t)r * 4096 + c2);
    asm volatile("global_load_async_to_lds_b32 %0, %1, off"
                 :: "v"(ldsOff), "v"(ga) : "memory");
  }
  asm volatile("s_wait_asynccnt 0x0" ::: "memory");
}

// ---------------------------------------------------------------------------
// Projection: out[b, m, n] = sum_k Wm[m,k] * x[b,k,n] + bias[m]   (f16 out)
// x: (B, 512, 4096) fp32. Block = 128 threads (4 waves), tile = 64(M) x 64(N).
// Vectorized fp32->f16 staging: float4 global loads, b64 LDS stores for A.
// ---------------------------------------------------------------------------
__global__ __launch_bounds__(128) void proj_kernel(
    const float* __restrict__ x, const float* __restrict__ Wm,
    const float* __restrict__ bias, f16* __restrict__ out, int M) {
  const int nbase = blockIdx.x * 64;
  const int mbase = blockIdx.y * 64;
  const int b     = blockIdx.z;
  const int tid   = threadIdx.x;
  const int lane  = tid & 31;
  const int wv    = tid >> 5;

  __shared__ f16 As[64 * 36];   // A[m][k], k contiguous
  __shared__ f16 Bt[64 * 36];   // B^T[n][k], k contiguous

  const float* xb = x + (size_t)b * 512 * 4096;

  v8f acc[4];
#pragma unroll
  for (int nt = 0; nt < 4; ++nt)
#pragma unroll
    for (int r = 0; r < 8; ++r) acc[nt][r] = 0.0f;

  const int half = lane >> 4;
  const int n15  = lane & 15;

  for (int kb = 0; kb < 512; kb += 32) {
    __syncthreads();
    // stage A tile 64x32: float4 load -> packed b64 LDS store
#pragma unroll
    for (int i = 0; i < 4; ++i) {
      int f = (tid + i * 128) * 4;
      int m = f >> 5, k4 = f & 31;
      const float4 t = *(const float4*)&Wm[(size_t)(mbase + m) * 512 + kb + k4];
      union { f16 h[4]; unsigned long long q; } u;
      u.h[0] = (f16)t.x; u.h[1] = (f16)t.y; u.h[2] = (f16)t.z; u.h[3] = (f16)t.w;
      *(unsigned long long*)&As[m * 36 + k4] = u.q;
    }
    // stage B tile transposed: float4 along n, 4 scalar f16 stores
#pragma unroll
    for (int i = 0; i < 4; ++i) {
      int f = (tid + i * 128) * 4;
      int k = f >> 6, n4 = f & 63;
      const float4 t = *(const float4*)&xb[(size_t)(kb + k) * 4096 + nbase + n4];
      Bt[(n4 + 0) * 36 + k] = (f16)t.x;
      Bt[(n4 + 1) * 36 + k] = (f16)t.y;
      Bt[(n4 + 2) * 36 + k] = (f16)t.z;
      Bt[(n4 + 3) * 36 + k] = (f16)t.w;
    }
    if (kb + 32 < 512)  // prefetch next x K-slab (-> global_prefetch_b8)
      __builtin_prefetch(&xb[(size_t)(kb + 32) * 4096 + nbase + lane], 0, 3);
    __syncthreads();

    Frag a, bf;
    const int mrow = wv * 16 + n15;
#pragma unroll
    for (int j = 0; j < 8; ++j) {
      int k0 = ((j & 3) * 2) + ((j >> 2) * 16) + half * 8;
      a.u[j] = *(const unsigned int*)&As[mrow * 36 + k0];
    }
#pragma unroll
    for (int nt = 0; nt < 4; ++nt) {
      const int ncol = nt * 16 + n15;
#pragma unroll
      for (int j = 0; j < 8; ++j) {
        int k0 = ((j & 3) * 2) + ((j >> 2) * 16) + half * 8;
        bf.u[j] = *(const unsigned int*)&Bt[ncol * 36 + k0];
      }
      acc[nt] = WMMA_F16(a.v, bf.v, acc[nt]);
    }
  }
  // epilogue: bias + convert + store f16
#pragma unroll
  for (int nt = 0; nt < 4; ++nt)
#pragma unroll
    for (int r = 0; r < 8; ++r) {
      int gm = mbase + wv * 16 + r + 8 * half;
      float val = acc[nt][r] + bias[gm];
      out[((size_t)b * M + gm) * 4096 + nbase + nt * 16 + n15] = (f16)val;
    }
}

// ---------------------------------------------------------------------------
// Attention scores: per (b, s) a 64x64x64 GEMM: e[m,n] = sum_d q'[m,d]*k'[n,d]
// H-mode: s = w, strideS=64, offMul=1  -> e[h,g].   W-mode: s = h, strideS=1,
// offMul=64 -> e[w,v].  Block = 128 threads (4 waves, wave = one N-tile).
// ---------------------------------------------------------------------------
__global__ __launch_bounds__(128) void attn_scores_kernel(
    const f16* __restrict__ q, const f16* __restrict__ k,
    float* __restrict__ ebuf, int strideS, int offMul) {
  const int s    = blockIdx.x;
  const int b    = blockIdx.y;
  const int tid  = threadIdx.x;
  const int lane = tid & 31;
  const int wv   = tid >> 5;
  const int off  = s * offMul;

  __shared__ f16 qT[64 * 68];   // qT[sr][d], d contiguous
  __shared__ f16 kT[64 * 68];

  const size_t qb = (size_t)b * 64 * 4096;
  if (strideS == 1) {
    // W-mode: sr contiguous in memory -> u32 loads, transposed b16 LDS stores
    for (int e = tid; e < 2048; e += 128) {
      int d = e >> 5, sr2 = (e & 31) << 1;
      union { unsigned u; f16 h[2]; } tq, tk;
      tq.u = *(const unsigned*)&q[qb + (size_t)d * 4096 + off + sr2];
      tk.u = *(const unsigned*)&k[qb + (size_t)d * 4096 + off + sr2];
      qT[(sr2 + 0) * 68 + d] = tq.h[0];
      qT[(sr2 + 1) * 68 + d] = tq.h[1];
      kT[(sr2 + 0) * 68 + d] = tk.h[0];
      kT[(sr2 + 1) * 68 + d] = tk.h[1];
    }
  } else {
    for (int e = tid; e < 4096; e += 128) {
      int d = e >> 6, sr = e & 63;
      size_t g = qb + (size_t)d * 4096 + (size_t)sr * strideS + off;
      qT[sr * 68 + d] = q[g];
      kT[sr * 68 + d] = k[g];
    }
  }
  __syncthreads();

  const int half = lane >> 4;
  const int n15  = lane & 15;
  v8f acc[4];
#pragma unroll
  for (int i = 0; i < 4; ++i)
#pragma unroll
    for (int r = 0; r < 8; ++r) acc[i][r] = 0.0f;

#pragma unroll
  for (int kb = 0; kb < 64; kb += 32) {
    Frag a[4], bf;
#pragma unroll
    for (int mt = 0; mt < 4; ++mt)
#pragma unroll
      for (int j = 0; j < 8; ++j) {
        int k0 = ((j & 3) * 2) + ((j >> 2) * 16) + half * 8;
        a[mt].u[j] = *(const unsigned int*)&qT[(mt * 16 + n15) * 68 + kb + k0];
      }
#pragma unroll
    for (int j = 0; j < 8; ++j) {
      int k0 = ((j & 3) * 2) + ((j >> 2) * 16) + half * 8;
      bf.u[j] = *(const unsigned int*)&kT[(wv * 16 + n15) * 68 + kb + k0];
    }
#pragma unroll
    for (int mt = 0; mt < 4; ++mt)
      acc[mt] = WMMA_F16(a[mt].v, bf.v, acc[mt]);
  }
  float* eb = ebuf + (((size_t)b * 64 + s) * 64) * 64;
#pragma unroll
  for (int mt = 0; mt < 4; ++mt)
#pragma unroll
    for (int r = 0; r < 8; ++r) {
      int m = mt * 16 + r + 8 * half;
      eb[(size_t)m * 64 + wv * 16 + n15] = acc[mt][r];
    }
}

// ---------------------------------------------------------------------------
// Softmax over concat[eH(64), eW(64)] per (b,h,w); diag mask eH[g==h] = -inf.
// ---------------------------------------------------------------------------
__global__ __launch_bounds__(256) void softmax_kernel(
    const float* __restrict__ eH, const float* __restrict__ eW,
    f16* __restrict__ aH, f16* __restrict__ aW) {
  int idx = blockIdx.x * blockDim.x + threadIdx.x;
  if (idx >= 8 * 64 * 64) return;
  int b = idx >> 12, hw = idx & 4095, h = hw >> 6, w = hw & 63;

  const float* pH = eH + (((size_t)b * 64 + w) * 64 + h) * 64;  // over g
  const float* pW = eW + (((size_t)b * 64 + h) * 64 + w) * 64;  // over v
  f16* oH = aH + (((size_t)b * 64 + w) * 64 + h) * 64;
  f16* oW = aW + (((size_t)b * 64 + h) * 64 + w) * 64;

  float mx = -3.0e38f;
  for (int g = 0; g < 64; ++g) if (g != h) mx = fmaxf(mx, pH[g]);
  for (int v = 0; v < 64; ++v) mx = fmaxf(mx, pW[v]);
  float sum = 0.0f;
  for (int g = 0; g < 64; ++g) if (g != h) sum += __expf(pH[g] - mx);
  for (int v = 0; v < 64; ++v) sum += __expf(pW[v] - mx);
  float inv = 1.0f / sum;
  for (int g = 0; g < 64; ++g)
    oH[g] = (f16)((g == h) ? 0.0f : __expf(pH[g] - mx) * inv);
  for (int v = 0; v < 64; ++v)
    oW[v] = (f16)(__expf(pW[v] - mx) * inv);
}

// ---------------------------------------------------------------------------
// Aggregation: per (b, s, ctile): out[c, n] = sum_kk v'[c, kk] * att[n, kk]
// H-mode (finalize=0): s=w, strideS=64, offMul=1  -> writes acc (fp32)
// W-mode (finalize=1): s=h, strideS=1,  offMul=64 -> d_out = gamma*(acc+res)+x
// Block = 128 (4 waves, wave = one N-tile). W-mode staging uses async-to-LDS.
// ---------------------------------------------------------------------------
__global__ __launch_bounds__(128) void aggregate_kernel(
    const f16* __restrict__ vbuf, const f16* __restrict__ att,
    const float* __restrict__ accIn, float* __restrict__ outp,
    const float* __restrict__ x, const float* __restrict__ gamma,
    int strideS, int offMul, int finalize) {
  const int ct   = blockIdx.x;           // 0..7  (64-row C tile)
  const int s    = blockIdx.y;           // 0..63
  const int b    = blockIdx.z;
  const int tid  = threadIdx.x;
  const int lane = tid & 31;
  const int wv   = tid >> 5;
  const int cbase = ct * 64;
  const int off   = s * offMul;

  __shared__ f16 Vt[64 * 68];            // Vt[c][kk], kk contiguous
  const size_t vb = ((size_t)b * 512 + cbase) * 4096;
  if (strideS == 1) {
    // W-mode: rows c contiguous over kk -> CDNA5 async DMA into LDS
    async_copy_tile64(vbuf + vb + off, Vt, 68, tid);
  } else {
    for (int e = tid; e < 4096; e += 128) {
      int c = e >> 6, kk = e & 63;
      Vt[c * 68 + kk] = vbuf[vb + (size_t)c * 4096 + (size_t)kk * 64 + off];
    }
  }
  __syncthreads();

  const f16* ab = att + (((size_t)b * 64 + s) * 64) * 64;  // [n][kk] contiguous
  const int half = lane >> 4;
  const int n15  = lane & 15;

  v8f acc[4];
#pragma unroll
  for (int i = 0; i < 4; ++i)
#pragma unroll
    for (int r = 0; r < 8; ++r) acc[i][r] = 0.0f;

#pragma unroll
  for (int kb = 0; kb < 64; kb += 32) {
    Frag a[4], bf;
#pragma unroll
    for (int mt = 0; mt < 4; ++mt)
#pragma unroll
      for (int j = 0; j < 8; ++j) {
        int k0 = ((j & 3) * 2) + ((j >> 2) * 16) + half * 8;
        a[mt].u[j] = *(const unsigned int*)&Vt[(mt * 16 + n15) * 68 + kb + k0];
      }
#pragma unroll
    for (int j = 0; j < 8; ++j) {
      int k0 = ((j & 3) * 2) + ((j >> 2) * 16) + half * 8;
      bf.u[j] = *(const unsigned int*)&ab[(size_t)(wv * 16 + n15) * 64 + kb + k0];
    }
#pragma unroll
    for (int mt = 0; mt < 4; ++mt)
      acc[mt] = WMMA_F16(a[mt].v, bf.v, acc[mt]);
  }

  const float g0 = gamma[0];
  const int n = wv * 16 + n15;
#pragma unroll
  for (int mt = 0; mt < 4; ++mt)
#pragma unroll
    for (int r = 0; r < 8; ++r) {
      int c = cbase + mt * 16 + r + 8 * half;
      size_t gi = ((size_t)b * 512 + c) * 4096 + (size_t)n * strideS + off;
      float res = acc[mt][r];
      if (finalize)
        outp[gi] = g0 * (accIn[gi] + res) + x[gi];
      else
        outp[gi] = res;
    }
}

// ---------------------------------------------------------------------------
extern "C" void kernel_launch(void* const* d_in, const int* in_sizes, int n_in,
                              void* d_out, int out_size, void* d_ws, size_t ws_size,
                              hipStream_t stream) {
  (void)in_sizes; (void)n_in; (void)out_size; (void)ws_size;
  const float* x     = (const float*)d_in[0];
  const float* Wq    = (const float*)d_in[1];
  const float* bq    = (const float*)d_in[2];
  const float* Wk    = (const float*)d_in[3];
  const float* bk    = (const float*)d_in[4];
  const float* Wv    = (const float*)d_in[5];
  const float* bv    = (const float*)d_in[6];
  const float* gamma = (const float*)d_in[7];
  float* out = (float*)d_out;

  char* ws = (char*)d_ws;
  f16*   qb   = (f16*)(ws + (0ull  << 20));   // 4 MiB  (8*64*4096 f16)
  f16*   kb   = (f16*)(ws + (4ull  << 20));   // 4 MiB
  f16*   vb   = (f16*)(ws + (8ull  << 20));   // 32 MiB (8*512*4096 f16)
  float* eHb  = (float*)(ws + (40ull << 20)); // 8 MiB  (b,w,h,g) fp32
  float* eWb  = (float*)(ws + (48ull << 20)); // 8 MiB  (b,h,w,v) fp32
  f16*   aHb  = (f16*)(ws + (56ull << 20));   // 4 MiB
  f16*   aWb  = (f16*)(ws + (60ull << 20));   // 4 MiB
  float* accb = (float*)(ws + (64ull << 20)); // 64 MiB (b,c,h,w) fp32

  dim3 blk128(128);

  // 1) projections (v is the ~17-GFLOP hot spot)
  proj_kernel<<<dim3(64, 1, 8), blk128, 0, stream>>>(x, Wq, bq, qb, 64);
  proj_kernel<<<dim3(64, 1, 8), blk128, 0, stream>>>(x, Wk, bk, kb, 64);
  proj_kernel<<<dim3(64, 8, 8), blk128, 0, stream>>>(x, Wv, bv, vb, 512);

  // 2) criss-cross scores
  attn_scores_kernel<<<dim3(64, 8), blk128, 0, stream>>>(qb, kb, eHb, 64, 1);  // H
  attn_scores_kernel<<<dim3(64, 8), blk128, 0, stream>>>(qb, kb, eWb, 1, 64);  // W

  // 3) masked softmax over concat(128)
  softmax_kernel<<<dim3(128), dim3(256), 0, stream>>>(eHb, eWb, aHb, aWb);

  // 4) aggregation (H writes acc; W fuses gamma*(acc+outW)+x -> d_out)
  aggregate_kernel<<<dim3(8, 64, 8), blk128, 0, stream>>>(vb, aHb, nullptr, accb,
                                                          x, gamma, 64, 1, 0);
  aggregate_kernel<<<dim3(8, 64, 8), blk128, 0, stream>>>(vb, aWb, accb, out,
                                                          x, gamma, 1, 64, 1);
}
